// BiLSTMCRF_58583353917653
// MI455X (gfx1250) — matrix operand order repaired and checked by reading, not compiled, not run on hardware
//
#include <hip/hip_runtime.h>
#include <hip/hip_bf16.h>

// ---------------------------------------------------------------------------
// BiLSTM-CRF forward NLL for MI455X (gfx1250, wave32, WMMA + async-to-LDS).
//
// Shapes: V=32000 E=256 H=256 4H=1024 HD=512 L=9 B=32 S=256, M=B*S=8192.
//
// Design:
//  * Dense projections: bf16 v_wmma_f32_16x16x32_bf16 GEMMs, f32 accumulate.
//    Each wave owns a 16x64 output strip (1 A fragment reused across 4
//    N-tiles / 4 accumulators) to halve A traffic and deepen the WMMA pipe.
//  * LSTM recurrence: persistent kernel, grid=2 (fwd/rev), 1024 threads
//    (32 waves) per direction. h lives in LDS as bf16 (WMMA A operand);
//    xp[t] (128KB/step) is double-buffered into LDS with
//    GLOBAL_LOAD_ASYNC_TO_LDS_B128 (ASYNCcnt) overlapping the WMMA phase.
//    LDS: 16KB h + 2x128KB xp = 272KB < 320KB/WGP (one WG per WGP here).
//  * Whh (bf16, 512KB/dir) re-read from global each step: resident in the
//    192MB L2; the recurrence is latency-bound, not bandwidth-bound.
//  * CRF: one wave, lane b = batch b, L=9 forward algorithm in registers,
//    wave shfl reduction to the scalar NLL.
// ---------------------------------------------------------------------------

typedef __attribute__((ext_vector_type(16))) __bf16 v16bf;
typedef __attribute__((ext_vector_type(8)))  __bf16 v8bf;
typedef __attribute__((ext_vector_type(8)))  float  v8f;
typedef int v4i_vs __attribute__((vector_size(16)));   // matches builtin proto

#define Bsz  32
#define Ssz  256
#define Esz  256
#define Hsz  256
#define G4   1024      // 4*H
#define HD2  512       // 2*H
#define Ltag 9
#define Mrow 8192      // B*S

// ---------------------- async global->LDS copy (CDNA5) ----------------------

__device__ __forceinline__ void async_b128(void* lds_dst, const void* gsrc) {
#if __has_builtin(__builtin_amdgcn_global_load_async_to_lds_b128)
  // Builtin proto (from toolchain diagnostic): (global v4i*, shared v4i*,
  // imm offset, imm cpol). Cast through integers: global generic == as1
  // bit-identical; low 32 bits of a generic LDS pointer are the LDS offset.
  __attribute__((address_space(1))) v4i_vs* g =
      (__attribute__((address_space(1))) v4i_vs*)(unsigned long long)(size_t)gsrc;
  __attribute__((address_space(3))) v4i_vs* l =
      (__attribute__((address_space(3))) v4i_vs*)(unsigned)(size_t)lds_dst;
  __builtin_amdgcn_global_load_async_to_lds_b128(g, l, 0, 0);
#else
  unsigned laddr = (unsigned)(size_t)lds_dst;
  asm volatile("global_load_async_to_lds_b128 %0, %1, off"
               :: "v"(laddr), "v"(gsrc) : "memory");
#endif
}

__device__ __forceinline__ void wait_async0() {
#if __has_builtin(__builtin_amdgcn_s_wait_asynccnt)
  __builtin_amdgcn_s_wait_asynccnt(0);
#else
  asm volatile("s_wait_asynccnt 0" ::: "memory");
#endif
}

// ----------------------------- helpers -------------------------------------

__global__ void f32_to_bf16_kernel(const float* __restrict__ in,
                                   __bf16* __restrict__ out, int n) {
  int i = blockIdx.x * blockDim.x + threadIdx.x;
  if (i < n) out[i] = (__bf16)in[i];
}

// x_bf16[b*S+s][e] = emb[ids[b*S+s]][e]
__global__ void embed_gather_kernel(const int* __restrict__ ids,
                                    const float* __restrict__ emb,
                                    __bf16* __restrict__ out) {
  int m = blockIdx.x;            // 0..8191
  int e = threadIdx.x;           // 0..255
  int tok = ids[m];
  out[(size_t)m * Esz + e] = (__bf16)emb[(size_t)tok * Esz + e];
}

// ------------------------- bf16 WMMA GEMM ----------------------------------
// C[M,N] (f32 rm) = A[M,K] (bf16 rm) * Bt[N,K]^T (bf16 rm).
// One 16(M) x 64(N) strip per wave: A fragment loaded once per 32-K chunk and
// reused by 4 WMMAs against 4 adjacent N-tiles.
__global__ void gemm_bf16_wmma_kernel(const __bf16* __restrict__ A,
                                      const __bf16* __restrict__ Bt,
                                      float* __restrict__ C,
                                      int M, int N, int K) {
  const int lane = threadIdx.x & 31;
  const int wid  = (blockIdx.x * blockDim.x + threadIdx.x) >> 5;
  const int ngrp = N >> 6;                       // groups of 4 N-tiles
  const int mt   = wid / ngrp;
  const int ng   = wid % ngrp;
  if (mt * 16 >= M) return;                      // uniform per wave

  const int row  = lane & 15;
  const int half = lane >> 4;

  const v8f vz = {0.f, 0.f, 0.f, 0.f, 0.f, 0.f, 0.f, 0.f};
  v8f acc0 = vz, acc1 = vz, acc2 = vz, acc3 = vz;

  const __bf16* arow = A + (size_t)(mt * 16 + row) * K + half * 16;
  // B fragment for N-tile q: lane(row,half) holds K = kc+half*16+[0..15],
  // N = ng*64 + q*16 + row  ->  row (ng*64 + q*16 + row) of Bt.
  const __bf16* brow0 = Bt + (size_t)(ng * 64 +  0 + row) * K + half * 16;
  const __bf16* brow1 = Bt + (size_t)(ng * 64 + 16 + row) * K + half * 16;
  const __bf16* brow2 = Bt + (size_t)(ng * 64 + 32 + row) * K + half * 16;
  const __bf16* brow3 = Bt + (size_t)(ng * 64 + 48 + row) * K + half * 16;

  for (int kc = 0; kc < K; kc += 32) {
    v16bf a;
    {
      v8bf lo = *(const v8bf*)(arow + kc);
      v8bf hi = *(const v8bf*)(arow + kc + 8);
#pragma unroll
      for (int i = 0; i < 8; ++i) { a[i] = lo[i]; a[i + 8] = hi[i]; }
    }
    v16bf b0 = *(const v16bf*)(brow0 + kc);
    v16bf b1 = *(const v16bf*)(brow1 + kc);
    v16bf b2 = *(const v16bf*)(brow2 + kc);
    v16bf b3 = *(const v16bf*)(brow3 + kc);
    if (kc + 32 < K) {
      __builtin_prefetch(arow  + kc + 32, 0, 3);
      __builtin_prefetch(brow0 + kc + 32, 0, 3);
      __builtin_prefetch(brow1 + kc + 32, 0, 3);
      __builtin_prefetch(brow2 + kc + 32, 0, 3);
      __builtin_prefetch(brow3 + kc + 32, 0, 3);
    }
    acc0 = __builtin_amdgcn_wmma_f32_16x16x32_bf16(false, a, false, b0, (short)0, acc0, false, false);
    acc1 = __builtin_amdgcn_wmma_f32_16x16x32_bf16(false, a, false, b1, (short)0, acc1, false, false);
    acc2 = __builtin_amdgcn_wmma_f32_16x16x32_bf16(false, a, false, b2, (short)0, acc2, false, false);
    acc3 = __builtin_amdgcn_wmma_f32_16x16x32_bf16(false, a, false, b3, (short)0, acc3, false, false);
  }

  // C/D layout: VGPR r -> M = half*8 + r, N = row (within each 16x16 tile)
  const int m0 = mt * 16 + half * 8;
  float* crow = C + (size_t)m0 * N + ng * 64 + row;
#pragma unroll
  for (int r = 0; r < 8; ++r) {
    crow[(size_t)r * N +  0] = acc0[r];
    crow[(size_t)r * N + 16] = acc1[r];
    crow[(size_t)r * N + 32] = acc2[r];
    crow[(size_t)r * N + 48] = acc3[r];
  }
}

// --------------------------- LSTM recurrence --------------------------------
// grid = 2 (blockIdx.x: 0=forward dir, 1=reverse dir), block = 1024 (32 waves).
// Per step: G[32,1024] = h_bf16[32,256] @ Whh^T via WMMA; xp[t] staged into
// LDS by async b128 copies double-buffered against the WMMA phase.
__global__ void __launch_bounds__(1024)
lstm_scan_kernel(const float* __restrict__ xp_f,   // [B*S, 4H] (M = b*S+s)
                 const float* __restrict__ xp_r,
                 const __bf16* __restrict__ whh_f,  // [4H, H] bf16 rm (== B^T)
                 const __bf16* __restrict__ whh_r,
                 const float* __restrict__ bias_f,  // [4H]
                 const float* __restrict__ bias_r,
                 float* __restrict__ out)           // [B, S, 2H]
{
  const int dir = blockIdx.x;
  const float*  xp   = dir ? xp_r  : xp_f;
  const __bf16* whh  = dir ? whh_r : whh_f;
  const float*  bias = dir ? bias_r : bias_f;

  __shared__ __bf16 hsh[Bsz * Hsz];                // 16 KB: h as WMMA A matrix
  __shared__ float  xpl[2][Bsz * G4];              // 2 x 128 KB: xp[t] stage

  const int tid  = threadIdx.x;
  const int lane = tid & 31;
  const int wave = tid >> 5;
  const int row  = lane & 15;
  const int half = lane >> 4;
  const int mt   = wave & 1;                       // batch tile
  const int nt0  = wave >> 1;                      // hidden 16-block, 0..15
  const int u    = nt0 * 16 + row;                 // hidden unit 0..255

  for (int i = tid; i < Bsz * Hsz; i += 1024) hsh[i] = (__bf16)0.0f;

  float c[8];
#pragma unroll
  for (int r = 0; r < 8; ++r) c[r] = 0.f;

  const float bi = bias[0 * Hsz + u];
  const float bf = bias[1 * Hsz + u];
  const float bg = bias[2 * Hsz + u];
  const float bo = bias[3 * Hsz + u];

  // Stage xp[t] (32*1024 f32 = 8192 16B chunks) into xpl[buf].
  // Thread tid copies chunks tid, tid+1024, ... (8 per thread).
  auto stage_xp = [&](int t, int buf) {
#pragma unroll
    for (int i = 0; i < 8; ++i) {
      const int cidx = i * 1024 + tid;             // 0..8191
      const int b    = cidx >> 8;                  // /256 chunks per batch row
      const int off4 = (cidx & 255) * 4;           // float offset within row
      const float* g = xp + ((size_t)b * Ssz + t) * G4 + off4;
      async_b128(&xpl[buf][(size_t)b * G4 + off4], g);
    }
  };

  const int t0 = dir ? (Ssz - 1) : 0;
  stage_xp(t0, 0);
  int cur = 0;

  __syncthreads();

  const v8f vzero = {0.f, 0.f, 0.f, 0.f, 0.f, 0.f, 0.f, 0.f};

  for (int step = 0; step < Ssz; ++step) {
    const int t = dir ? (Ssz - 1 - step) : step;

    v8f acc0 = vzero, acc1 = vzero, acc2 = vzero, acc3 = vzero;

#pragma unroll 2
    for (int kc = 0; kc < Hsz; kc += 32) {
      v16bf a;
      {
        const __bf16* ap = &hsh[(size_t)(mt * 16 + row) * Hsz + kc + half * 16];
        v8bf lo = *(const v8bf*)ap;
        v8bf hi = *(const v8bf*)(ap + 8);
#pragma unroll
        for (int i = 0; i < 8; ++i) { a[i] = lo[i]; a[i + 8] = hi[i]; }
      }
      const __bf16* bbase = whh + (size_t)(nt0 * 16 + row) * Hsz + kc + half * 16;
      v16bf b0 = *(const v16bf*)(bbase + (size_t)(0 * Hsz) * Hsz);
      v16bf b1 = *(const v16bf*)(bbase + (size_t)(1 * Hsz) * Hsz);
      v16bf b2 = *(const v16bf*)(bbase + (size_t)(2 * Hsz) * Hsz);
      v16bf b3 = *(const v16bf*)(bbase + (size_t)(3 * Hsz) * Hsz);
      acc0 = __builtin_amdgcn_wmma_f32_16x16x32_bf16(false, a, false, b0, (short)0, acc0, false, false);
      acc1 = __builtin_amdgcn_wmma_f32_16x16x32_bf16(false, a, false, b1, (short)0, acc1, false, false);
      acc2 = __builtin_amdgcn_wmma_f32_16x16x32_bf16(false, a, false, b2, (short)0, acc2, false, false);
      acc3 = __builtin_amdgcn_wmma_f32_16x16x32_bf16(false, a, false, b3, (short)0, acc3, false, false);
    }

    wait_async0();     // this wave's xp[t] chunks are in LDS
    __syncthreads();   // all waves' chunks visible; h(t-1) reads done

    const float* xrow = &xpl[cur][0];
#pragma unroll
    for (int r = 0; r < 8; ++r) {
      const int bidx = mt * 16 + half * 8 + r;     // C/D layout: M = half*8+r
      const size_t xb = (size_t)bidx * G4;
      float gi = acc0[r] + xrow[xb + 0 * Hsz + u] + bi;
      float gf = acc1[r] + xrow[xb + 1 * Hsz + u] + bf;
      float gg = acc2[r] + xrow[xb + 2 * Hsz + u] + bg;
      float go = acc3[r] + xrow[xb + 3 * Hsz + u] + bo;
      float si = 1.f / (1.f + __expf(-gi));
      float sf = 1.f / (1.f + __expf(-gf));
      float so = 1.f / (1.f + __expf(-go));
      c[r] = sf * c[r] + si * tanhf(gg);
      float h = so * tanhf(c[r]);
      hsh[(size_t)bidx * Hsz + u] = (__bf16)h;
      out[((size_t)bidx * Ssz + t) * HD2 + dir * Hsz + u] = h;
    }

    if (step + 1 < Ssz) {
      const int tn = dir ? (Ssz - 2 - step) : (step + 1);
      stage_xp(tn, cur ^ 1);   // overlaps next step's WMMA phase
    }

    __syncthreads();   // h(t) visible to all waves
    cur ^= 1;
  }
}

// ----------------------------- emissions -----------------------------------
// em[m][l] = x2[m][:] . w_out[l][:] + b_out[l]   (L=9 is too skinny for WMMA)
__global__ void emissions_kernel(const float* __restrict__ x,
                                 const float* __restrict__ w,
                                 const float* __restrict__ b,
                                 float* __restrict__ em) {
  int idx = blockIdx.x * blockDim.x + threadIdx.x;
  if (idx >= Mrow * Ltag) return;
  int m = idx / Ltag, l = idx % Ltag;
  const float* xr = x + (size_t)m * HD2;
  const float* wr = w + (size_t)l * HD2;
  float s = b[l];
#pragma unroll 8
  for (int k = 0; k < HD2; ++k) s += xr[k] * wr[k];
  em[idx] = s;
}

// ------------------------------- CRF NLL ------------------------------------
__global__ void crf_nll_kernel(const float* __restrict__ em,     // [B,S,L]
                               const int* __restrict__ labels,   // [B,S]
                               const int* __restrict__ mask,     // [B,S]
                               const float* __restrict__ start,
                               const float* __restrict__ endv,
                               const float* __restrict__ trans,  // [L,L]
                               float* __restrict__ out) {
  const int b = threadIdx.x;                       // one lane per batch elem
  float alpha[Ltag], nxt[Ltag];
  const float* e0 = em + (size_t)b * Ssz * Ltag;
  int prev = labels[b * Ssz + 0];
  float num = start[prev] + e0[prev];
  for (int j = 0; j < Ltag; ++j) alpha[j] = start[j] + e0[j];

  for (int t = 1; t < Ssz; ++t) {
    const float* et = em + ((size_t)b * Ssz + t) * Ltag;
    const int   tg  = labels[b * Ssz + t];
    const int   mk  = mask[b * Ssz + t];
    const float mf  = (float)mk;
    num += (trans[prev * Ltag + tg] + et[tg]) * mf;
    if (mk > 0) prev = tg;
    for (int j = 0; j < Ltag; ++j) {
      float mx = -3.4e38f;
      for (int i = 0; i < Ltag; ++i) mx = fmaxf(mx, alpha[i] + trans[i * Ltag + j]);
      float s = 0.f;
      for (int i = 0; i < Ltag; ++i) s += __expf(alpha[i] + trans[i * Ltag + j] - mx);
      nxt[j] = mx + __logf(s) + et[j];
    }
    if (mk > 0)
      for (int j = 0; j < Ltag; ++j) alpha[j] = nxt[j];
  }
  num += endv[prev];
  float mx = -3.4e38f;
  for (int j = 0; j < Ltag; ++j) mx = fmaxf(mx, alpha[j] + endv[j]);
  float s = 0.f;
  for (int j = 0; j < Ltag; ++j) s += __expf(alpha[j] + endv[j] - mx);
  float val = num - (mx + __logf(s));
  // wave32 reduction
  for (int off = 16; off > 0; off >>= 1) val += __shfl_down(val, off, 32);
  if (b == 0) out[0] = -val;
}

// ------------------------------- launcher -----------------------------------

static inline char* take(char*& p, size_t bytes) {
  char* r = p;
  p += (bytes + 255) & ~(size_t)255;
  return r;
}

extern "C" void kernel_launch(void* const* d_in, const int* in_sizes, int n_in,
                              void* d_out, int out_size, void* d_ws, size_t ws_size,
                              hipStream_t stream) {
  const int*   ids   = (const int*)  d_in[0];
  const int*   amask = (const int*)  d_in[1];
  const int*   labels= (const int*)  d_in[2];
  const float* emb   = (const float*)d_in[3];
  const float* wih0f = (const float*)d_in[4];
  const float* whh0f = (const float*)d_in[5];
  const float* b0f   = (const float*)d_in[6];
  const float* wih0r = (const float*)d_in[7];
  const float* whh0r = (const float*)d_in[8];
  const float* b0r   = (const float*)d_in[9];
  const float* wih1f = (const float*)d_in[10];
  const float* whh1f = (const float*)d_in[11];
  const float* b1f   = (const float*)d_in[12];
  const float* wih1r = (const float*)d_in[13];
  const float* whh1r = (const float*)d_in[14];
  const float* b1r   = (const float*)d_in[15];
  const float* wout  = (const float*)d_in[16];
  const float* bout  = (const float*)d_in[17];
  const float* startt= (const float*)d_in[18];
  const float* endt  = (const float*)d_in[19];
  const float* trans = (const float*)d_in[20];
  float* out = (float*)d_out;

  // ---- workspace partition (~120 MB) ----
  char* p = (char*)d_ws;
  __bf16* x_bf16   = (__bf16*)take(p, (size_t)Mrow * Esz * 2);   // 4 MB
  float*  x1       = (float*) take(p, (size_t)Mrow * HD2 * 4);   // 16 MB
  __bf16* x1_bf16  = (__bf16*)take(p, (size_t)Mrow * HD2 * 2);   // 8 MB
  float*  x2       = (float*) take(p, (size_t)Mrow * HD2 * 4);   // 16 MB
  float*  xp_a     = (float*) take(p, (size_t)Mrow * G4  * 4);   // 32 MB
  float*  xp_b     = (float*) take(p, (size_t)Mrow * G4  * 4);   // 32 MB
  float*  emis     = (float*) take(p, (size_t)Mrow * Ltag * 4);
  __bf16* wih0f_b  = (__bf16*)take(p, (size_t)G4 * Esz * 2);
  __bf16* wih0r_b  = (__bf16*)take(p, (size_t)G4 * Esz * 2);
  __bf16* whh0f_b  = (__bf16*)take(p, (size_t)G4 * Hsz * 2);
  __bf16* whh0r_b  = (__bf16*)take(p, (size_t)G4 * Hsz * 2);
  __bf16* wih1f_b  = (__bf16*)take(p, (size_t)G4 * HD2 * 2);
  __bf16* wih1r_b  = (__bf16*)take(p, (size_t)G4 * HD2 * 2);
  __bf16* whh1f_b  = (__bf16*)take(p, (size_t)G4 * Hsz * 2);
  __bf16* whh1r_b  = (__bf16*)take(p, (size_t)G4 * Hsz * 2);

  // ---- weight conversions f32 -> bf16 ----
  auto cvt = [&](const float* src, __bf16* dst, int n) {
    f32_to_bf16_kernel<<<(n + 255) / 256, 256, 0, stream>>>(src, dst, n);
  };
  cvt(wih0f, wih0f_b, G4 * Esz);
  cvt(wih0r, wih0r_b, G4 * Esz);
  cvt(whh0f, whh0f_b, G4 * Hsz);
  cvt(whh0r, whh0r_b, G4 * Hsz);
  cvt(wih1f, wih1f_b, G4 * HD2);
  cvt(wih1r, wih1r_b, G4 * HD2);
  cvt(whh1f, whh1f_b, G4 * Hsz);
  cvt(whh1r, whh1r_b, G4 * Hsz);

  // ---- embedding gather (f32 -> bf16) ----
  embed_gather_kernel<<<Mrow, Esz, 0, stream>>>(ids, emb, x_bf16);

  // ---- layer 0: input projections (WMMA) + recurrence ----
  {
    const int waves  = (Mrow / 16) * (G4 / 64);          // 8192 waves
    const int blocks = waves / 8;                        // 8 waves / block
    gemm_bf16_wmma_kernel<<<blocks, 256, 0, stream>>>(x_bf16, wih0f_b, xp_a,
                                                      Mrow, G4, Esz);
    gemm_bf16_wmma_kernel<<<blocks, 256, 0, stream>>>(x_bf16, wih0r_b, xp_b,
                                                      Mrow, G4, Esz);
  }
  lstm_scan_kernel<<<2, 1024, 0, stream>>>(xp_a, xp_b, whh0f_b, whh0r_b,
                                           b0f, b0r, x1);

  // ---- layer 1 ----
  cvt(x1, x1_bf16, Mrow * HD2);
  {
    const int waves  = (Mrow / 16) * (G4 / 64);
    const int blocks = waves / 8;
    gemm_bf16_wmma_kernel<<<blocks, 256, 0, stream>>>(x1_bf16, wih1f_b, xp_a,
                                                      Mrow, G4, HD2);
    gemm_bf16_wmma_kernel<<<blocks, 256, 0, stream>>>(x1_bf16, wih1r_b, xp_b,
                                                      Mrow, G4, HD2);
  }
  lstm_scan_kernel<<<2, 1024, 0, stream>>>(xp_a, xp_b, whh1f_b, whh1r_b,
                                           b1f, b1r, x2);

  // ---- emissions + CRF ----
  emissions_kernel<<<(Mrow * Ltag + 255) / 256, 256, 0, stream>>>(x2, wout,
                                                                  bout, emis);
  crf_nll_kernel<<<1, 32, 0, stream>>>(emis, labels, amask, startt, endt,
                                       trans, out);
}
